// WordAttention_50964081934379
// MI455X (gfx1250) — matrix operand order repaired and checked by reading, not compiled, not run on hardware
//
#include <hip/hip_runtime.h>

// Problem constants (from reference)
#define NROWS  65536   // BATCH*MAXLEN_SENTENCE*MAXLEN_WORD rows of the GEMM
#define DDIM   512     // FEATURE_DIM (GEMM K)
#define ADIM   256     // ATTEN_SIZE  (GEMM N)
#define WLEN   64      // MAXLEN_WORD
#define NSENTW 1024    // BATCH*MAXLEN_SENTENCE
#define EPSF   1e-7f

typedef __attribute__((ext_vector_type(16))) __bf16     v16bf;
typedef __attribute__((ext_vector_type(8)))  float      v8f;
typedef __attribute__((ext_vector_type(8)))  unsigned   v8u;

// ---- helpers -------------------------------------------------------------

// pack two f32 -> one dword of two bf16 (RNE). Prefer HW V_CVT_PK_BF16_F32.
#if __has_builtin(__builtin_amdgcn_cvt_pk_bf16_f32)
__device__ __forceinline__ unsigned pack_bf16_rne(float lo, float hi) {
  auto p = __builtin_amdgcn_cvt_pk_bf16_f32(lo, hi);
  static_assert(sizeof(p) == 4, "cvt_pk_bf16_f32 must return 32 bits");
  unsigned u;
  __builtin_memcpy(&u, &p, 4);
  return u;
}
#else
__device__ __forceinline__ unsigned pack_bf16_rne(float lo, float hi) {
  unsigned a = __float_as_uint(lo), b = __float_as_uint(hi);
  a += 0x7fffu + ((a >> 16) & 1u);
  b += 0x7fffu + ((b >> 16) & 1u);
  return (a >> 16) | (b & 0xffff0000u);
}
#endif

// hardware tanh (V_TANH_F32 trans op on gfx1250), fallback to libm
#if __has_builtin(__builtin_amdgcn_tanhf)
#define FAST_TANH(x) __builtin_amdgcn_tanhf(x)
#elif __has_builtin(__builtin_amdgcn_tanh_f32)
#define FAST_TANH(x) __builtin_amdgcn_tanh_f32(x)
#else
#define FAST_TANH(x) tanhf(x)
#endif

// order-preserving float <-> unsigned key (for atomicMax on floats)
__device__ __forceinline__ unsigned fkey(float f) {
  unsigned u = __float_as_uint(f);
  return (u & 0x80000000u) ? ~u : (u | 0x80000000u);
}
__device__ __forceinline__ float funkey(unsigned k) {
  unsigned u = (k & 0x80000000u) ? (k & 0x7fffffffu) : ~k;
  return __uint_as_float(u);
}

// ---- kernel 1: build bf16 B-fragments for V_WMMA_F32_16X16X32_BF16 -------
// B tile (32x16, KxN), wave32 fragment: lane = N (0..15 / 16..31 halves),
// lanes 0-15 hold K=0..15, lanes 16-31 hold K=16..31, 2 bf16 per dword.
// Global layout: dword index = ((kc*16 + t)*32 + lane)*8 + q
__global__ __launch_bounds__(256) void wa_prep(const float* __restrict__ kern,
                                               unsigned* __restrict__ bfrag,
                                               unsigned* __restrict__ maxkey) {
  int idx = blockIdx.x * 256 + threadIdx.x;          // 0..65535
  if (idx == 0) *maxkey = fkey(-INFINITY);
  int q    = idx & 7;
  int lane = (idx >> 3) & 31;
  int t    = (idx >> 8) & 15;
  int kc   = idx >> 12;
  int k = (kc << 5) + ((lane >> 4) << 4) + (q << 1); // K of low half of dword
  int n = (t << 4) + (lane & 15);
  float lo = kern[k * ADIM + n];
  float hi = kern[(k + 1) * ADIM + n];
  bfrag[idx] = pack_bf16_rne(lo, hi);
}

// ---- kernel 2: fused GEMM + tanh + dot(u) -> kv, global max --------------
// Block = 512 threads (16 waves). Wave w owns rows m0..m0+15, all 256 cols.
__global__ __launch_bounds__(512) void wa_gemm_kv(const float* __restrict__ inputs,
                                                  const float* __restrict__ bias,
                                                  const float* __restrict__ uctx,
                                                  const uint4* __restrict__ bfrag,
                                                  float* __restrict__ kv,
                                                  unsigned* __restrict__ maxkey) {
  __shared__ uint4 sB[1024];                 // one k-chunk: 16 tiles * 32 lanes * 32B = 16 KB
  const int tid  = threadIdx.x;
  const int lane = tid & 31;
  const int wv   = tid >> 5;
  const int m0   = (blockIdx.x * 16 + wv) << 4;            // first row of this wave's tile
  const int r    = m0 + (lane & 15);
  const float* __restrict__ arow = inputs + (size_t)r * DDIM;
  const int khalf = (lane >> 4) << 4;                      // A frag: lanes 16-31 offset K by 16

  v8f c[16] = {};                                          // 16 C tiles (16x16 f32) = 128 VGPRs

  for (int kc = 0; kc < 16; ++kc) {
    __syncthreads();                                       // protect sB reuse
    sB[tid]       = bfrag[(kc << 10) + tid];
    sB[tid + 512] = bfrag[(kc << 10) + tid + 512];
    __syncthreads();

    // A fragment 16x32 bf16: lane half reads K window [kb, kb+16)
    const int kb = (kc << 5) + khalf;
    float4 f0 = *(const float4*)(arow + kb);
    float4 f1 = *(const float4*)(arow + kb + 4);
    float4 g0 = *(const float4*)(arow + kb + 8);
    float4 g1 = *(const float4*)(arow + kb + 12);
    v8u au;
    au[0] = pack_bf16_rne(f0.x, f0.y); au[1] = pack_bf16_rne(f0.z, f0.w);
    au[2] = pack_bf16_rne(f1.x, f1.y); au[3] = pack_bf16_rne(f1.z, f1.w);
    au[4] = pack_bf16_rne(g0.x, g0.y); au[5] = pack_bf16_rne(g0.z, g0.w);
    au[6] = pack_bf16_rne(g1.x, g1.y); au[7] = pack_bf16_rne(g1.z, g1.w);
    v16bf a = __builtin_bit_cast(v16bf, au);

    // B fragments: explicit one-tile-ahead double buffer so the backend can
    // keep a ds_load pair in flight behind each WMMA (s_wait_dscnt >= 1)
    const int bbase = lane << 1;                            // uint4 index of (t=0, lane)
    uint4 pb0 = sB[bbase];
    uint4 pb1 = sB[bbase + 1];
#pragma unroll
    for (int t = 0; t < 16; ++t) {
      uint4 cb0 = pb0, cb1 = pb1;
      if (t < 15) {
        pb0 = sB[bbase + ((t + 1) << 6)];
        pb1 = sB[bbase + ((t + 1) << 6) + 1];
      }
      v8u bu = { cb0.x, cb0.y, cb0.z, cb0.w, cb1.x, cb1.y, cb1.z, cb1.w };
      c[t] = __builtin_amdgcn_wmma_f32_16x16x32_bf16(
                 false, a, false, __builtin_bit_cast(v16bf, bu),
                 (short)0, c[t], false, false);
    }
  }

  // epilogue: kv[m] = sum_n tanh(c[m][n] + bias[n]) * u[n]
  // C layout: VGPR j holds M = j (lanes 0-15) / j+8 (lanes 16-31), N = (lane&15) + 16*t
  float p[8] = {0.f, 0.f, 0.f, 0.f, 0.f, 0.f, 0.f, 0.f};
#pragma unroll
  for (int t = 0; t < 16; ++t) {
    int col = (t << 4) + (lane & 15);
    float bb = bias[col];
    float uu = uctx[col];
#pragma unroll
    for (int j = 0; j < 8; ++j)
      p[j] += FAST_TANH(c[t][j] + bb) * uu;
  }
#pragma unroll
  for (int j = 0; j < 8; ++j)
    for (int off = 8; off > 0; off >>= 1)
      p[j] += __shfl_xor(p[j], off, 16);                   // reduce within 16-lane halves

  if ((lane & 15) == 0) {
    int base = m0 + ((lane >> 4) << 3);
    float lmax = -INFINITY;
#pragma unroll
    for (int j = 0; j < 8; ++j) { kv[base + j] = p[j]; lmax = fmaxf(lmax, p[j]); }
    atomicMax(maxkey, fkey(lmax));
  }
}

// ---- kernel 3: softmax over W + weighted sum over W (bandwidth bound) ----
__global__ __launch_bounds__(256) void wa_softmax_reduce(const float* __restrict__ inputs,
                                                         const float* __restrict__ mask,
                                                         const float* __restrict__ kv,
                                                         const unsigned* __restrict__ maxkey,
                                                         float* __restrict__ outVec,
                                                         float* __restrict__ outProb) {
  __shared__ float sE[WLEN];
  __shared__ float sDen;
  const int n   = blockIdx.x;
  const int tid = threadIdx.x;
  const float maxv = funkey(*maxkey);

  if (tid < WLEN) {
    float e = __expf(kv[n * WLEN + tid] - maxv) + EPSF;
    e *= mask[n * WLEN + tid];
    sE[tid] = e;
  }
  __syncthreads();
  if (tid == 0) {
    float s = 0.f;
    for (int w = 0; w < WLEN; ++w) s += sE[w];
    sDen = s + EPSF;
  }
  __syncthreads();
  if (tid < WLEN) {
    float pr = sE[tid] / sDen;
    sE[tid] = pr;
    outProb[n * WLEN + tid] = pr;
  }
  __syncthreads();

  const float2* __restrict__ inp = (const float2*)(inputs + (size_t)n * WLEN * DDIM);
  float2 acc = make_float2(0.f, 0.f);
  for (int w = 0; w < WLEN; ++w) {
    float2 v = inp[w * (DDIM / 2) + tid];
    float pr = sE[w];
    acc.x += pr * v.x;
    acc.y += pr * v.y;
  }
  ((float2*)outVec)[(size_t)n * (DDIM / 2) + tid] = acc;
}

// ---- launch --------------------------------------------------------------
extern "C" void kernel_launch(void* const* d_in, const int* in_sizes, int n_in,
                              void* d_out, int out_size, void* d_ws, size_t ws_size,
                              hipStream_t stream) {
  const float* inputs = (const float*)d_in[0];   // (65536, 512)
  const float* mask   = (const float*)d_in[1];   // (65536, 1) per (n,w)
  const float* kern   = (const float*)d_in[2];   // (512, 256)
  const float* bias   = (const float*)d_in[3];   // (256,)
  const float* uctx   = (const float*)d_in[4];   // (256, 1)

  float* outVec  = (float*)d_out;                  // sentence_vector: 1024*512
  float* outProb = outVec + (size_t)NSENTW * DDIM; // prob_reshape: 1024*64

  // workspace: [0..255] max key, then 256 KB B-fragments, then 256 KB kv
  unsigned* maxkey = (unsigned*)d_ws;
  unsigned* bfrag  = (unsigned*)((char*)d_ws + 256);
  float*    kv     = (float*)((char*)d_ws + 256 + (size_t)65536 * 4);

  wa_prep<<<256, 256, 0, stream>>>(kern, bfrag, maxkey);
  wa_gemm_kv<<<256, 512, 0, stream>>>(inputs, bias, uctx, (const uint4*)bfrag, kv, maxkey);
  wa_softmax_reduce<<<NSENTW, 256, 0, stream>>>(inputs, mask, kv, maxkey, outVec, outProb);
}
